// ConvLSTM_46505905881490
// MI455X (gfx1250) — matrix operand order repaired and checked by reading, not compiled, not run on hardware
//
#include <hip/hip_runtime.h>

typedef __attribute__((ext_vector_type(16))) _Float16 v16h;
typedef __attribute__((ext_vector_type(8)))  _Float16 v8h;
typedef __attribute__((ext_vector_type(8)))  float    v8f;
typedef __attribute__((ext_vector_type(4)))  unsigned int u32x4;
typedef __attribute__((ext_vector_type(8)))  int i32x8;
typedef __attribute__((ext_vector_type(4)))  int i32x4;

#define B_  8
#define T_  6
#define C_  28
#define HW_ 16384
#define N_  131072

// ---- LDS weight image: f16 B-fragment regions (offsets in halves) ----
#define O_RED 0          // reduce:  O=48,  Kp=32  -> 3 blocks  (1536 halves)
#define O_IH1 1536       // w_ih1:   O=256, Kp=64  -> 32 blocks (16384)
#define O_HH1 17920      // w_hh1:   O=256, K=64   -> 32 blocks (16384)
#define O_FUS 34304      // fused:   O=512, K=64   -> 64 blocks (32768)
#define O_HH2 67072      // w_hh2:   O=512, K=128  -> 128 blocks(65536)
#define W_HALVES 132608

// ---- f32 bias/vec region (offsets in floats, after halves) ----
#define F_B1 0           // 256: b_ih1+b_hh1
#define F_B2 256         // 512: b_ih2+b_hh2+w_ih2@conv1_b
#define F_BR 768         // 64:  reduce_b (padded)
#define F_V  832         // 128: head_w @ conv2_w
#define F_S  960         // 1:   head_b + head_w . conv2_b
#define F_TOT 964

#define IMG_BYTES (W_HALVES*2 + F_TOT*4)   // 269072, 16B aligned, = 33634 * 8
#define IMG_QW    (IMG_BYTES/8)            // 33634 8-byte elements for TDM

// ---- per-wave activation staging (row-major f16), offsets in halves ----
#define A_X   0          // [16][32]
#define A_IN  512        // [16][64]  (cols 48..63 stay zero)
#define A_H1  1536       // [16][64]
#define A_H2  2560       // [16][128]
#define ACT_HALVES 4608
#define ACT_BYTES (ACT_HALVES*2)
#define LDS_TOTAL (IMG_BYTES + 4*ACT_BYTES)   // 305936 B <= 320KB/WGP

// ws layout: [0,IMG_BYTES) = LDS image; then f32 w_fused temp [512*64]
#define WS_FUS_OFF IMG_BYTES

__device__ __forceinline__ float fsig(float x){ return 1.0f/(1.0f + __expf(-x)); }
__device__ __forceinline__ float ftanh(float x){ float e = __expf(2.0f*x); return (e-1.0f)/(e+1.0f); }

// A-fragment (16x32 f16, M x K) from row-major LDS [16][ldk]
__device__ __forceinline__ v16h ldsA(const _Float16* base, int ldk, int kt, int lane){
  int row = lane & 15, kb = (lane >> 4) * 16;
  const _Float16* p = base + row*ldk + kt*32;
  union { v16h v; v8h h[2]; } u;
  u.h[0] = *(const v8h*)(p + kb);        // K = kb .. kb+7
  u.h[1] = *(const v8h*)(p + 8 + kb);    // K = 8+kb .. 8+kb+7
  return u.v;
}
// B-fragment (32x16 f16, K x N) from pre-swizzled block: 512 contiguous halves
__device__ __forceinline__ v16h ldsB(const _Float16* blk, int lane){
  const _Float16* p = blk + lane*16;
  union { v16h v; v8h h[2]; } u;
  u.h[0] = *(const v8h*)p;
  u.h[1] = *(const v8h*)(p + 8);
  return u.v;
}
#define WMMA(a,b,c) __builtin_amdgcn_wmma_f32_16x16x32_f16(false,(a),false,(b),(short)0,(c),false,false)

// ---------- prep 1: fused weights / biases / head vector (f32) ----------
__global__ void convlstm_prep1(const float* __restrict__ w_ih2, const float* __restrict__ conv1_w,
                               const float* __restrict__ conv1_b,
                               const float* __restrict__ b_ih1, const float* __restrict__ b_hh1,
                               const float* __restrict__ b_ih2, const float* __restrict__ b_hh2,
                               const float* __restrict__ conv2_w, const float* __restrict__ conv2_b,
                               const float* __restrict__ head_w, const float* __restrict__ head_b,
                               const float* __restrict__ reduce_b, char* __restrict__ ws)
{
  int idx = blockIdx.x*blockDim.x + threadIdx.x;
  float* fimg = (float*)(ws + W_HALVES*2);
  float* wfus = (float*)(ws + WS_FUS_OFF);
  if (idx < 32768) {                       // w_fused[o][k] = sum_m w_ih2[o][m]*conv1_w[m][k]
    int o = idx >> 6, k = idx & 63;
    float s = 0.f;
    for (int m = 0; m < 64; ++m) s += w_ih2[o*64 + m] * conv1_w[m*64 + k];
    wfus[idx] = s;
  } else if (idx < 33024) {                // b1
    int i = idx - 32768; fimg[F_B1 + i] = b_ih1[i] + b_hh1[i];
  } else if (idx < 33536) {                // b2 (fused)
    int o = idx - 33024;
    float s = b_ih2[o] + b_hh2[o];
    for (int m = 0; m < 64; ++m) s += w_ih2[o*64 + m] * conv1_b[m];
    fimg[F_B2 + o] = s;
  } else if (idx < 33664) {                // v = head_w @ conv2_w
    int j = idx - 33536;
    float s = 0.f;
    for (int k = 0; k < 128; ++k) s += head_w[k] * conv2_w[k*128 + j];
    fimg[F_V + j] = s;
  } else if (idx == 33664) {               // scalar offset
    float s = head_b[0];
    for (int k = 0; k < 128; ++k) s += head_w[k] * conv2_b[k];
    fimg[F_S] = s; fimg[F_S+1] = 0.f; fimg[F_S+2] = 0.f; fimg[F_S+3] = 0.f;
  } else if (idx < 33729) {                // reduce bias (padded to 64)
    int i = idx - 33665; fimg[F_BR + i] = (i < 48) ? reduce_b[i] : 0.f;
  }
}

// ---------- prep 2: swizzle all weights into f16 WMMA B-fragment layout ----------
__global__ void convlstm_prep2(const float* __restrict__ reduce_w,
                               const float* __restrict__ w_ih1,
                               const float* __restrict__ w_hh1,
                               const float* __restrict__ w_hh2,
                               char* __restrict__ ws)
{
  int idx = blockIdx.x*blockDim.x + threadIdx.x;
  if (idx >= W_HALVES) return;
  _Float16* wh = (_Float16*)ws;
  const float* wfus = (const float*)(ws + WS_FUS_OFF);
  int r, nK;
  if (idx < O_IH1)      { r = idx;          nK = 1; }
  else if (idx < O_HH1) { r = idx - O_IH1;  nK = 2; }
  else if (idx < O_FUS) { r = idx - O_HH1;  nK = 2; }
  else if (idx < O_HH2) { r = idx - O_FUS;  nK = 2; }
  else                  { r = idx - O_HH2;  nK = 4; }
  int blk = r >> 9, rem = r & 511, l = rem >> 4, h = rem & 15;
  int ot = blk / nK, kt = blk % nK;          // block = ot*nK + kt
  int o = ot*16 + (l & 15);                  // output column (N)
  int k = kt*32 + (l >> 4)*16 + h;           // K index
  float val;
  if (idx < O_IH1)      val = (k < 28) ? reduce_w[o*28 + k] : 0.f;
  else if (idx < O_HH1) val = (k < 48) ? w_ih1[o*48 + k] : 0.f;
  else if (idx < O_FUS) val = w_hh1[o*64 + k];
  else if (idx < O_HH2) val = wfus[o*64 + k];
  else                  val = w_hh2[o*128 + k];
  wh[idx] = (_Float16)val;
}

// ---------- main: 1 WG = 4 waves = 64 pixels, full T-loop in-register/LDS ----------
__global__ __launch_bounds__(128, 1) void convlstm_main(const float* __restrict__ x,
                                                        const char* __restrict__ ws,
                                                        float* __restrict__ out)
{
  extern __shared__ char smem[];
  _Float16* wh = (_Float16*)smem;
  float*    fb = (float*)(smem + W_HALVES*2);
  const int tid  = threadIdx.x;
  const int wave = tid >> 5, lane = tid & 31;
  _Float16* actw = (_Float16*)(smem + IMG_BYTES + wave*ACT_BYTES);
  _Float16* aX  = actw + A_X;
  _Float16* aIn = actw + A_IN;
  _Float16* aH1 = actw + A_H1;
  _Float16* aH2 = actw + A_H2;

  // ---- stage 269KB weight image global -> LDS via Tensor Data Mover ----
#if __has_builtin(__builtin_amdgcn_tensor_load_to_lds)
  if (wave == 0) {
    unsigned long long ga = (unsigned long long)(const void*)ws;
    // D# group0: count=1 | lds_addr=0 | global_addr[56:0] | type=2
    u32x4 g0;
    g0[0] = 1u;                                            // count=1, user descriptor
    g0[1] = 0u;                                            // lds_addr = 0 (WG LDS base)
    g0[2] = (unsigned int)(ga & 0xFFFFFFFFu);              // global_addr[31:0]
    g0[3] = (unsigned int)((ga >> 32) & 0x01FFFFFFu)       // global_addr[56:32]
          | (2u << 30);                                    // type = 2 ("image")
    // D# group1: 1-D tensor of 33634 x 8B; tile = whole tensor
    i32x8 g1 = {};
    g1[0] = (3 << 16);                                     // data_size = 8 bytes
    g1[1] = (int)((IMG_QW & 0xFFFF) << 16);                // tensor_dim0[15:0]  (bits 63:48)
    g1[2] = (int)((IMG_QW >> 16) & 0xFFFF);                // tensor_dim0[31:16] (bits 79:64)
    g1[3] = (int)((IMG_QW & 0xFFFF) << 16);                // tile_dim0 (bits 127:112)
    g1[4] = 0;                                             // tile_dim1/2 = 0 (1-D tile)
    g1[5] = (int)IMG_QW;                                   // tensor_dim0_stride[31:0]
    g1[6] = 0;
    g1[7] = 0;
    i32x4 gz = {};
#if __clang_major__ >= 23
    i32x8 gz8 = {};
    __builtin_amdgcn_tensor_load_to_lds(g0, g1, gz, gz, gz8, 0);
#else
    __builtin_amdgcn_tensor_load_to_lds(g0, g1, gz, gz, 0);
#endif
  }
#else
  {
    const uint4* src = (const uint4*)ws;
    uint4* dst = (uint4*)smem;
    for (int i = tid; i < IMG_BYTES/16; i += 128) dst[i] = src[i];
  }
#endif

  // zero activation staging (overlaps with TDM DMA); h=0 at t=0, pads stay zero
  {
    uint4 z; z.x = z.y = z.z = z.w = 0u;
    uint4* a = (uint4*)actw;
    for (int i = lane; i < ACT_BYTES/16; i += 32) a[i] = z;
  }

#if __has_builtin(__builtin_amdgcn_tensor_load_to_lds)
  if (wave == 0) __builtin_amdgcn_s_wait_tensorcnt(0);
#endif
  __syncthreads();

  const int pix0 = blockIdx.x*64 + wave*16;
  const int row = lane & 15;
  const int hi  = lane >> 4;
  const int col = row;                 // D/C: element v -> row v+hi*8, col lane&15
  const int p  = pix0 + row;
  const int bb = p >> 14;              // p / (H*W)
  const int hw = p & (HW_ - 1);
  const int cbase = hi * 14;

  v8f vz = {};
  v8f c1s[4], c2s[8];
#pragma unroll
  for (int j = 0; j < 4; ++j) c1s[j] = vz;
#pragma unroll
  for (int j = 0; j < 8; ++j) c2s[j] = vz;

  for (int t = 0; t < T_; ++t) {
    // ---- gather x channels, apply u/v denormalization, stage as f16 A rows ----
#pragma unroll
    for (int cc = 0; cc < 14; ++cc) {
      int c = cbase + cc;
      float val = x[(((size_t)bb*T_ + t)*C_ + c)*HW_ + hw];
      float sc = (c == 11) ? 0.1f  : ((c == 12) ? 0.12f : 1.0f);
      float sh = (c == 11) ? 0.01f : ((c == 12) ? 0.02f : 0.0f);
      aX[row*32 + c] = (_Float16)(val*sc + sh);
    }
    // prefetch next timestep's channel gather into cache (global_prefetch_b8)
    if (t + 1 < T_) {
#pragma unroll
      for (int cc = 0; cc < 14; ++cc) {
        int c = cbase + cc;
        __builtin_prefetch((const void*)&x[(((size_t)bb*T_ + (t+1))*C_ + c)*HW_ + hw], 0, 0);
      }
    }

    // ---- reduce 28(->32) -> 48 ----
    {
      v16h ax = ldsA(aX, 32, 0, lane);
#pragma unroll
      for (int ot = 0; ot < 3; ++ot) {
        v8f acc = vz + fb[F_BR + ot*16 + col];
        acc = WMMA(ax, ldsB(wh + O_RED + ot*512, lane), acc);
#pragma unroll
        for (int v = 0; v < 8; ++v)
          aIn[(v + hi*8)*64 + ot*16 + col] = (_Float16)acc[v];
      }
    }

    // ---- LSTM1: gates[256] = in48 @ w_ih1^T + h1 @ w_hh1^T + b ----
    {
      v16h ain0 = ldsA(aIn, 64, 0, lane), ain1 = ldsA(aIn, 64, 1, lane);
      v16h ah0  = ldsA(aH1, 64, 0, lane), ah1  = ldsA(aH1, 64, 1, lane); // old h1 in regs
#pragma unroll
      for (int j = 0; j < 4; ++j) {
        v8f acc[4];
#pragma unroll
        for (int g = 0; g < 4; ++g) acc[g] = vz + fb[F_B1 + g*64 + j*16 + col];
#pragma unroll
        for (int g = 0; g < 4; ++g) {
          int ot = g*4 + j;
          acc[g] = WMMA(ain0, ldsB(wh + O_IH1 + (ot*2+0)*512, lane), acc[g]);
          acc[g] = WMMA(ain1, ldsB(wh + O_IH1 + (ot*2+1)*512, lane), acc[g]);
          acc[g] = WMMA(ah0 , ldsB(wh + O_HH1 + (ot*2+0)*512, lane), acc[g]);
          acc[g] = WMMA(ah1 , ldsB(wh + O_HH1 + (ot*2+1)*512, lane), acc[g]);
        }
#pragma unroll
        for (int v = 0; v < 8; ++v) {
          float iv = fsig(acc[0][v]), fv = fsig(acc[1][v]);
          float gv = ftanh(acc[2][v]), ov = fsig(acc[3][v]);
          float cn = fv*c1s[j][v] + iv*gv;
          c1s[j][v] = cn;
          aH1[(v + hi*8)*64 + j*16 + col] = (_Float16)(ov*ftanh(cn));
        }
      }
    }

    // ---- LSTM2 (conv1 fused into input path): gates[512] ----
    {
      v16h b10 = ldsA(aH1, 64, 0, lane), b11 = ldsA(aH1, 64, 1, lane);  // new h1
      v16h ah2[4];
#pragma unroll
      for (int kt = 0; kt < 4; ++kt) ah2[kt] = ldsA(aH2, 128, kt, lane); // old h2 in regs
#pragma unroll
      for (int j = 0; j < 8; ++j) {
        v8f acc[4];
#pragma unroll
        for (int g = 0; g < 4; ++g) acc[g] = vz + fb[F_B2 + g*128 + j*16 + col];
#pragma unroll
        for (int g = 0; g < 4; ++g) {
          int ot = g*8 + j;
          acc[g] = WMMA(b10, ldsB(wh + O_FUS + (ot*2+0)*512, lane), acc[g]);
          acc[g] = WMMA(b11, ldsB(wh + O_FUS + (ot*2+1)*512, lane), acc[g]);
#pragma unroll
          for (int kt = 0; kt < 4; ++kt)
            acc[g] = WMMA(ah2[kt], ldsB(wh + O_HH2 + (ot*4+kt)*512, lane), acc[g]);
        }
#pragma unroll
        for (int v = 0; v < 8; ++v) {
          float iv = fsig(acc[0][v]), fv = fsig(acc[1][v]);
          float gv = ftanh(acc[2][v]), ov = fsig(acc[3][v]);
          float cn = fv*c2s[j][v] + iv*gv;
          c2s[j][v] = cn;
          aH2[(v + hi*8)*128 + j*16 + col] = (_Float16)(ov*ftanh(cn));
        }
      }
    }
  }

  // ---- folded conv2+head: out = h2_final . v + s ----
  if (lane < 16) {
    float sum = fb[F_S];
    const _Float16* hrow = aH2 + lane*128;
#pragma unroll 8
    for (int k = 0; k < 128; ++k) sum += (float)hrow[k] * fb[F_V + k];
    out[pix0 + lane] = sum;
  }
}

extern "C" void kernel_launch(void* const* d_in, const int* in_sizes, int n_in,
                              void* d_out, int out_size, void* d_ws, size_t ws_size,
                              hipStream_t stream)
{
  const float* x        = (const float*)d_in[0];
  const float* reduce_w = (const float*)d_in[1];
  const float* reduce_b = (const float*)d_in[2];
  const float* w_ih1    = (const float*)d_in[3];
  const float* w_hh1    = (const float*)d_in[4];
  const float* b_ih1    = (const float*)d_in[5];
  const float* b_hh1    = (const float*)d_in[6];
  const float* conv1_w  = (const float*)d_in[7];
  const float* conv1_b  = (const float*)d_in[8];
  const float* w_ih2    = (const float*)d_in[9];
  const float* w_hh2    = (const float*)d_in[10];
  const float* b_ih2    = (const float*)d_in[11];
  const float* b_hh2    = (const float*)d_in[12];
  const float* conv2_w  = (const float*)d_in[13];
  const float* conv2_b  = (const float*)d_in[14];
  const float* head_w   = (const float*)d_in[15];
  const float* head_b   = (const float*)d_in[16];
  char*  ws  = (char*)d_ws;
  float* out = (float*)d_out;

  convlstm_prep1<<<(33729 + 127)/128, 128, 0, stream>>>(
      w_ih2, conv1_w, conv1_b, b_ih1, b_hh1, b_ih2, b_hh2,
      conv2_w, conv2_b, head_w, head_b, reduce_b, ws);
  convlstm_prep2<<<W_HALVES/128, 128, 0, stream>>>(reduce_w, w_ih1, w_hh1, w_hh2, ws);

  hipFuncSetAttribute(reinterpret_cast<const void*>(convlstm_main),
                      hipFuncAttributeMaxDynamicSharedMemorySize, LDS_TOTAL);
  convlstm_main<<<N_/64, 128, LDS_TOTAL, stream>>>(x, ws, out);
}
